// Net_separate_11390253269734
// MI455X (gfx1250) — matrix operand order repaired and checked by reading, not compiled, not run on hardware
//
#include <hip/hip_runtime.h>
#include <hip/hip_bf16.h>

#define NN  100000
#define EE  1600000
#define NBB 20000
#define HH  32

typedef __attribute__((ext_vector_type(16))) _Float16 v16h;
typedef __attribute__((ext_vector_type(8)))  _Float16 v8h;
typedef __attribute__((ext_vector_type(8)))  float    v8f;

// ---------------- degree / norm precompute ----------------

__global__ void k_init_one(float* __restrict__ deg, int n) {
    int i = blockIdx.x * blockDim.x + threadIdx.x;
    if (i < n) deg[i] = 1.0f;           // self loop contributes 1 to every degree
}

__global__ void k_deg_count(const int* __restrict__ ei, float* __restrict__ deg, int e) {
    int t = blockIdx.x * blockDim.x + threadIdx.x;
    if (t < e) atomicAdd(&deg[ei[e + t]], 1.0f);   // dst = edge_index[1]
}

__global__ void k_rsqrt_inplace(float* __restrict__ d, int n) {
    int i = blockIdx.x * blockDim.x + threadIdx.x;
    if (i < n) d[i] = rsqrtf(d[i]);     // deg >= 1 always (self loop)
}

__global__ void k_edge_norm(const int* __restrict__ ei, const float* __restrict__ dinv,
                            float* __restrict__ nrm, int e) {
    int t = blockIdx.x * blockDim.x + threadIdx.x;
    if (t < e) nrm[t] = dinv[ei[t]] * dinv[ei[e + t]];
}

// ---------------- encoder MLPs (VALU; weights stay wave-uniform -> s_load) ----------------

__global__ void k_enc_boundary(const float* __restrict__ x, const float* __restrict__ y,
                               const int* __restrict__ bidx,
                               const float* __restrict__ W1, const float* __restrict__ b1,
                               const float* __restrict__ W2, const float* __restrict__ b2,
                               _Float16* __restrict__ h16, int nb) {
    int t = blockIdx.x * blockDim.x + threadIdx.x;
    if (t >= nb) return;
    int node = bidx[t];
    float in0 = x[2 * node], in1 = x[2 * node + 1], in2 = y[node];
    float hid[HH];
#pragma unroll
    for (int j = 0; j < HH; ++j) {
        float v = b1[j] + in0 * W1[j] + in1 * W1[HH + j] + in2 * W1[2 * HH + j];
        hid[j] = v > 0.0f ? v : 0.0f;
    }
    float acc[HH];
#pragma unroll
    for (int j = 0; j < HH; ++j) acc[j] = b2[j];
    for (int k = 0; k < HH; ++k) {
        float hk = hid[k];
#pragma unroll
        for (int j = 0; j < HH; ++j) acc[j] += hk * W2[k * HH + j];
    }
    _Float16* op = h16 + (size_t)node * HH;
#pragma unroll
    for (int j = 0; j < HH; ++j)
        op[j] = (_Float16)(acc[j] > 0.0f ? acc[j] : 0.0f);   // relu(h) from reference
}

__global__ void k_enc_interior(const float* __restrict__ x,
                               const int* __restrict__ iidx,
                               const float* __restrict__ W1, const float* __restrict__ b1,
                               const float* __restrict__ W2, const float* __restrict__ b2,
                               _Float16* __restrict__ h16, int ni) {
    int t = blockIdx.x * blockDim.x + threadIdx.x;
    if (t >= ni) return;
    int node = iidx[t];
    float in0 = x[2 * node], in1 = x[2 * node + 1];
    float hid[HH];
#pragma unroll
    for (int j = 0; j < HH; ++j) {
        float v = b1[j] + in0 * W1[j] + in1 * W1[HH + j];
        hid[j] = v > 0.0f ? v : 0.0f;
    }
    float acc[HH];
#pragma unroll
    for (int j = 0; j < HH; ++j) acc[j] = b2[j];
    for (int k = 0; k < HH; ++k) {
        float hk = hid[k];
#pragma unroll
        for (int j = 0; j < HH; ++j) acc[j] += hk * W2[k * HH + j];
    }
    _Float16* op = h16 + (size_t)node * HH;
#pragma unroll
    for (int j = 0; j < HH; ++j)
        op[j] = (_Float16)(acc[j] > 0.0f ? acc[j] : 0.0f);
}

// ---------------- weight repack into WMMA B-lane layout ----------------
// Tile n0 = 16*tile. Lane l: rows K = (l<16?0:16)+i, col = n0+(l&15), i=0..15.
// Stored so the GEMM kernel loads a contiguous v16h per lane.

__global__ void k_prep_w(const float* __restrict__ W, _Float16* __restrict__ wp) {
    int t = threadIdx.x;            // launched with 64 threads, 1 block
    int tile = t >> 5, lane = t & 31;
    int kb = (lane >> 4) << 4;      // 0 or 16
    int n = tile * 16 + (lane & 15);
#pragma unroll
    for (int i = 0; i < 16; ++i)
        wp[tile * 512 + lane * 16 + i] = (_Float16)W[(kb + i) * HH + n];
}

// ---------------- dense transform ht = h16 @ W via v_wmma_f32_16x16x32_f16 ----------------

__global__ __launch_bounds__(256) void k_gemm_wmma(const _Float16* __restrict__ h16,
                                                   const _Float16* __restrict__ wp,
                                                   float* __restrict__ ht, int nrows) {
    int wave = threadIdx.x >> 5;
    int lane = threadIdx.x & 31;
    int rowBase = (blockIdx.x * 8 + wave) * 16;
    if (rowBase >= nrows) return;   // wave-uniform: EXEC stays all-ones for WMMA

    // A: 16x32 f16. lanes 0-15 -> K {0..15}; lanes 16-31 -> K {16..31}
    int row   = rowBase + (lane & 15);
    int kbase = (lane >> 4) << 4;   // 0 or 16
    const _Float16* ap = h16 + (size_t)row * HH + kbase;
    v8h lo = *(const v8h*)ap;
    v8h hi = *(const v8h*)(ap + 8);
    v16h a;
#pragma unroll
    for (int i = 0; i < 8; ++i) { a[i] = lo[i]; a[8 + i] = hi[i]; }

    v16h b0 = *(const v16h*)(wp + lane * 16);
    v16h b1 = *(const v16h*)(wp + 512 + lane * 16);

    v8f z = {};
    v8f c0 = __builtin_amdgcn_wmma_f32_16x16x32_f16(false, a, false, b0, (short)0, z, false, false);
    v8f c1 = __builtin_amdgcn_wmma_f32_16x16x32_f16(false, a, false, b1, (short)0, z, false, false);

    // C layout: VGPR v -> M = v + (lane<16 ? 0 : 8), N = lane&15
    int mOff = (lane >> 4) << 3;
    int col  = lane & 15;
#pragma unroll
    for (int v = 0; v < 8; ++v) {
        ht[(size_t)(rowBase + mOff + v) * HH + col]      = c0[v];
        ht[(size_t)(rowBase + mOff + v) * HH + 16 + col] = c1[v];
    }
}

// ---------------- aggregation: zero, edge scatter-add, epilogues ----------------

__global__ void k_zero(float* __restrict__ p, int n) {
    int i = blockIdx.x * blockDim.x + threadIdx.x;
    if (i < n) p[i] = 0.0f;
}

// 8 threads per edge, 4 columns each (float4 gather + 4 f32 atomics). agg is L2-resident.
__global__ void k_scatter(const int* __restrict__ ei, const float* __restrict__ nrm,
                          const float* __restrict__ ht, float* __restrict__ agg, int e) {
    int t = blockIdx.x * blockDim.x + threadIdx.x;
    int edge = t >> 3;
    if (edge >= e) return;
    int part = t & 7;
    int s = ei[edge];
    int d = ei[e + edge];
    float w = nrm[edge];
    const float4 v = *(const float4*)(ht + (size_t)s * HH + part * 4);
    float* ap = agg + (size_t)d * HH + part * 4;
    atomicAdd(ap + 0, v.x * w);
    atomicAdd(ap + 1, v.y * w);
    atomicAdd(ap + 2, v.z * w);
    atomicAdd(ap + 3, v.w * w);
}

// layers 1,2 epilogue: add self loop (ht[i]*dinv^2) + bias, relu, re-encode to f16
__global__ void k_bias_relu(const float* __restrict__ agg, const float* __restrict__ ht,
                            const float* __restrict__ dinv, const float* __restrict__ b,
                            _Float16* __restrict__ h16, int total) {
    int t = blockIdx.x * blockDim.x + threadIdx.x;
    if (t >= total) return;
    int n = t >> 5, c = t & 31;
    float di = dinv[n];
    float v = agg[t] + ht[t] * di * di + b[c];
    h16[t] = (_Float16)(v > 0.0f ? v : 0.0f);
}

// layer 3 epilogue fused with head: out = relu(agg + self + b3) @ Wf + bf
__global__ void k_final(const float* __restrict__ agg, const float* __restrict__ ht,
                        const float* __restrict__ dinv, const float* __restrict__ b3,
                        const float* __restrict__ Wf, const float* __restrict__ bf,
                        float* __restrict__ out, int n) {
    int i = blockIdx.x * blockDim.x + threadIdx.x;
    if (i >= n) return;
    float d2 = dinv[i] * dinv[i];
    float acc = bf[0];
    const float* ag = agg + (size_t)i * HH;
    const float* hp = ht + (size_t)i * HH;
#pragma unroll
    for (int c = 0; c < HH; ++c) {
        float v = ag[c] + hp[c] * d2 + b3[c];
        v = v > 0.0f ? v : 0.0f;
        acc += v * Wf[c];
    }
    out[i] = acc;
}

// ---------------- host side ----------------

static inline size_t align256(size_t x) { return (x + 255) & ~(size_t)255; }

extern "C" void kernel_launch(void* const* d_in, const int* in_sizes, int n_in,
                              void* d_out, int out_size, void* d_ws, size_t ws_size,
                              hipStream_t stream) {
    const float* x    = (const float*)d_in[0];
    const float* y    = (const float*)d_in[1];
    const int*   ei   = (const int*)d_in[2];
    const int*   bidx = (const int*)d_in[3];
    const int*   iidx = (const int*)d_in[4];
    const float* Wb1 = (const float*)d_in[5];  const float* bb1 = (const float*)d_in[6];
    const float* Wb2 = (const float*)d_in[7];  const float* bb2 = (const float*)d_in[8];
    const float* Wi1 = (const float*)d_in[9];  const float* bi1 = (const float*)d_in[10];
    const float* Wi2 = (const float*)d_in[11]; const float* bi2 = (const float*)d_in[12];
    const float* Wc[3] = { (const float*)d_in[13], (const float*)d_in[15], (const float*)d_in[17] };
    const float* bc[3] = { (const float*)d_in[14], (const float*)d_in[16], (const float*)d_in[18] };
    const float* Wf = (const float*)d_in[19];
    const float* bf = (const float*)d_in[20];
    float* out = (float*)d_out;

    char* ws = (char*)d_ws;
    size_t off = 0;
    float*    dinv = (float*)(ws + off);    off += align256((size_t)NN * 4);
    float*    nrm  = (float*)(ws + off);    off += align256((size_t)EE * 4);
    _Float16* h16  = (_Float16*)(ws + off); off += align256((size_t)NN * HH * 2);
    float*    ht   = (float*)(ws + off);    off += align256((size_t)NN * HH * 4);
    float*    agg  = (float*)(ws + off);    off += align256((size_t)NN * HH * 4);
    _Float16* wp   = (_Float16*)(ws + off); off += align256(1024 * 2);
    (void)ws_size; (void)n_in; (void)in_sizes; (void)out_size;

    const int B = 256;
    // degree / dinv / per-edge norm (reused by all 3 conv layers)
    k_init_one<<<(NN + B - 1) / B, B, 0, stream>>>(dinv, NN);
    k_deg_count<<<(EE + B - 1) / B, B, 0, stream>>>(ei, dinv, EE);
    k_rsqrt_inplace<<<(NN + B - 1) / B, B, 0, stream>>>(dinv, NN);
    k_edge_norm<<<(EE + B - 1) / B, B, 0, stream>>>(ei, dinv, nrm, EE);

    // encoders -> relu -> h16
    k_enc_boundary<<<(NBB + B - 1) / B, B, 0, stream>>>(x, y, bidx, Wb1, bb1, Wb2, bb2, h16, NBB);
    k_enc_interior<<<((NN - NBB) + B - 1) / B, B, 0, stream>>>(x, iidx, Wi1, bi1, Wi2, bi2, h16, NN - NBB);

    const int gemmBlocks = (NN / 16 + 7) / 8;        // 6250 tiles, 8 waves/block
    const int totHF = NN * HH;

    for (int l = 0; l < 3; ++l) {
        k_prep_w<<<1, 64, 0, stream>>>(Wc[l], wp);
        k_gemm_wmma<<<gemmBlocks, 256, 0, stream>>>(h16, wp, ht, NN);
        k_zero<<<(totHF + B - 1) / B, B, 0, stream>>>(agg, totHF);
        k_scatter<<<((EE * 8) + B - 1) / B, B, 0, stream>>>(ei, nrm, ht, agg, EE);
        if (l < 2) {
            k_bias_relu<<<(totHF + B - 1) / B, B, 0, stream>>>(agg, ht, dinv, bc[l], h16, totHF);
        } else {
            k_final<<<(NN + B - 1) / B, B, 0, stream>>>(agg, ht, dinv, bc[l], Wf, bf, out, NN);
        }
    }
}
